// RhnLin_53231824667308
// MI455X (gfx1250) — matrix operand order repaired and checked by reading, not compiled, not run on hardware
//
#include <hip/hip_runtime.h>
#include <hip/hip_bf16.h>

#define DI __device__ __forceinline__

typedef __attribute__((ext_vector_type(16))) __bf16 v16bf;
typedef __attribute__((ext_vector_type(8)))  __bf16 v8bf;
typedef __attribute__((ext_vector_type(8)))  float  v8f;
typedef __attribute__((ext_vector_type(4)))  float  v4f;
typedef __attribute__((ext_vector_type(4)))  int    v4i;

typedef __attribute__((address_space(1))) v4i as1_v4i;
typedef __attribute__((address_space(3))) v4i as3_v4i;

constexpr int B = 32, T = 1024, I = 1024, H = 1024, L = 3;
constexpr int NT = H / 16;        // 64 n-tiles of 16 columns
constexpr int RHN_BLOCKS = 8;     // 8 blocks * 8 waves = 64 waves = NT
constexpr int PITCH = 1032;       // LDS row pitch in bf16 (2064B -> 4-bank row skew)

#if defined(__has_builtin)
#if __has_builtin(__builtin_amdgcn_global_load_async_to_lds_b128) && \
    __has_builtin(__builtin_amdgcn_s_wait_asynccnt)
#define HAVE_ASYNC_LDS 1
#endif
#endif
#ifndef HAVE_ASYNC_LDS
#define HAVE_ASYNC_LDS 0
#endif

// ---------------------------------------------------------------------------
// Fragment loaders (CDNA5 wave32 WMMA 16x16x32 bf16 layouts, ISA 7.12.2)
// A 16x32 (MxK), 16-bit: lane m = lane&15, kb = lane>>4.
//   elems 0..7  -> K = k0 + kb*8 + 0..7        (16B contiguous)
//   elems 8..15 -> K = k0 + 16 + kb*8 + 0..7   (16B contiguous)
// B 32x16 (KxN), 16-bit: lane n = lane&15, kb = lane>>4.
//   elems 0..7  -> K = k0 + kb*8 + 0..7, elems 8..15 -> K = k0 + 16 + kb*8 + 0..7
// C/D 16x16 f32: VGPR v -> M = v + (lane>>4)*8, N = lane&15.
// ---------------------------------------------------------------------------

DI v16bf load_a_bf16(const __bf16* __restrict__ row, int k0, int lane) {
    int kb = (lane >> 4) & 1;
    v8bf lo = *(const v8bf*)(row + k0 + kb * 8);
    v8bf hi = *(const v8bf*)(row + k0 + 16 + kb * 8);
    return __builtin_shufflevector(lo, hi, 0, 1, 2, 3, 4, 5, 6, 7,
                                   8, 9, 10, 11, 12, 13, 14, 15);
}

DI v16bf load_b_bf16(const __bf16* __restrict__ nrow, int k0, int lane) {
    int kb = (lane >> 4) & 1;
    v8bf lo = *(const v8bf*)(nrow + k0 + kb * 8);
    v8bf hi = *(const v8bf*)(nrow + k0 + 16 + kb * 8);
    return __builtin_shufflevector(lo, hi, 0, 1, 2, 3, 4, 5, 6, 7,
                                   8, 9, 10, 11, 12, 13, 14, 15);
}

DI v8f wmma_bf16(v16bf a, v16bf b, v8f c) {
    return __builtin_amdgcn_wmma_f32_16x16x32_bf16(false, a, false, b,
                                                   (short)0, c, false, false);
}

DI float sigmoidf_fast(float x) { return 1.0f / (1.0f + __expf(-x)); }

// Cooperative copy of `nrows` rows of `row_elems` bf16 from global into LDS
// (pitch PITCH), using the CDNA5 async global->LDS engine when available.
DI void stage_rows_to_lds(__bf16* lds, const __bf16* gsrc, int nrows, int row_elems) {
    int chunks_per_row = (row_elems * 2) / 16;     // 16B chunks
    int total = nrows * chunks_per_row;
    for (int i = threadIdx.x; i < total; i += blockDim.x) {
        int r = i / chunks_per_row;
        int c = i % chunks_per_row;
        const char* src = (const char*)(gsrc + (size_t)r * row_elems) + c * 16;
        char* dst = (char*)(lds + (size_t)r * PITCH) + c * 16;
#if HAVE_ASYNC_LDS
        __builtin_amdgcn_global_load_async_to_lds_b128(
            (as1_v4i*)(void*)const_cast<char*>(src),
            (as3_v4i*)dst, 0, 0);
#else
        *(v8bf*)dst = *(const v8bf*)src;
#endif
    }
#if HAVE_ASYNC_LDS
    __builtin_amdgcn_s_wait_asynccnt(0);
#endif
    __syncthreads();
}

// ---------------------------------------------------------------------------
// Prep kernels
// ---------------------------------------------------------------------------

// x_bf[j][k] = (bf16)x[j][k], j = b*T+t  (one-time conversion; ~64MB, L2-warm)
__global__ void k_prep_x(const float* __restrict__ x, __bf16* __restrict__ xb) {
    size_t i4 = (size_t)blockIdx.x * blockDim.x + threadIdx.x;
    size_t tot4 = (size_t)B * T * I / 4;
    if (i4 >= tot4) return;
    v4f v = *(const v4f*)(x + i4 * 4);
    __bf16* o = xb + i4 * 4;
    o[0] = (__bf16)v[0]; o[1] = (__bf16)v[1];
    o[2] = (__bf16)v[2]; o[3] = (__bf16)v[3];
}

// wT[g][n][k] = w_g[k*H + n]  (bf16), g in {h,t,c}
__global__ void k_prep_w(const float* __restrict__ w_h,
                         const float* __restrict__ w_t,
                         const float* __restrict__ w_c,
                         __bf16* __restrict__ wT) {
    size_t idx = (size_t)blockIdx.x * blockDim.x + threadIdx.x;
    size_t tot = (size_t)3 * H * I;
    if (idx >= tot) return;
    int g = (int)(idx / ((size_t)H * I));
    size_t r = idx % ((size_t)H * I);
    int n = (int)(r / I);
    int k = (int)(r % I);
    const float* w = (g == 0) ? w_h : (g == 1) ? w_t : w_c;
    wT[idx] = (__bf16)w[(size_t)k * H + n];
}

// Rb[l][g][n][k] = R_g[l][n][k]  (bf16; B[k][n]=R[n][k] needs no transpose)
__global__ void k_prep_R(const float* __restrict__ Rh,
                         const float* __restrict__ Rt,
                         const float* __restrict__ Rc,
                         __bf16* __restrict__ Rb) {
    size_t idx = (size_t)blockIdx.x * blockDim.x + threadIdx.x;
    size_t tot = (size_t)L * 3 * H * H;
    if (idx >= tot) return;
    int l = (int)(idx / ((size_t)3 * H * H));
    size_t r = idx % ((size_t)3 * H * H);
    int g = (int)(r / ((size_t)H * H));
    size_t e = r % ((size_t)H * H);
    const float* R = (g == 0) ? Rh : (g == 1) ? Rt : Rc;
    Rb[idx] = (__bf16)R[(size_t)l * H * H + e];
}

__global__ void k_prep_s(const float* __restrict__ s0,
                         float* __restrict__ s_a,
                         __bf16* __restrict__ s_bf_a,
                         int* __restrict__ bar) {
    int idx = blockIdx.x * blockDim.x + threadIdx.x;
    if (idx < B * H) {
        float v = s0[idx];
        s_a[idx] = v;
        s_bf_a[idx] = (__bf16)v;
    }
    if (idx < 64) bar[idx] = 0;
}

// ---------------------------------------------------------------------------
// Phase 1: U[g][t][b][h] = (X @ w_g)[b*T+t][h] as bf16, f32 accumulation.
// Workgroup = (gate, 32-row block). A row-block staged once in LDS (async),
// shared by 8 waves; each wave covers 8 n-tiles.
// ---------------------------------------------------------------------------
__global__ void __launch_bounds__(256) k_proj(const __bf16* __restrict__ xb,
                                              const __bf16* __restrict__ wT,
                                              __bf16* __restrict__ U) {
    __shared__ __bf16 a_lds[32 * PITCH];

    const int rblocks = (B * T) / 32;                  // 1024
    int g    = blockIdx.x / rblocks;
    int rb   = blockIdx.x % rblocks;
    int row0 = rb * 32;

    stage_rows_to_lds(a_lds, xb + (size_t)row0 * I, 32, I);

    int wave = threadIdx.x >> 5;
    int lane = threadIdx.x & 31;
    const __bf16* a0r = a_lds + (size_t)(lane & 15) * PITCH;
    const __bf16* a1r = a_lds + (size_t)(16 + (lane & 15)) * PITCH;

#pragma unroll 1
    for (int nn = 0; nn < 8; ++nn) {
        int nt = wave * 8 + nn;
        const __bf16* nrow =
            wT + (size_t)g * H * I + (size_t)(nt * 16 + (lane & 15)) * I;

        v8f acc0 = {};
        v8f acc1 = {};
#pragma unroll 4
        for (int k0 = 0; k0 < I; k0 += 32) {
            v16bf bf = load_b_bf16(nrow, k0, lane);
            v16bf a0 = load_a_bf16(a0r, k0, lane);
            v16bf a1 = load_a_bf16(a1r, k0, lane);
            acc0 = wmma_bf16(a0, bf, acc0);
            acc1 = wmma_bf16(a1, bf, acc1);
        }

        int c = nt * 16 + (lane & 15);
#pragma unroll
        for (int v = 0; v < 8; ++v) {
            int j0 = row0 + ((lane >> 4) & 1) * 8 + v;  // M-tile 0
            int j1 = j0 + 16;                           // M-tile 1
            int b0 = j0 / T, t0 = j0 % T;
            int b1 = j1 / T, t1 = j1 % T;
            U[(((size_t)g * T + t0) * B + b0) * H + c] = (__bf16)acc0[v];
            U[(((size_t)g * T + t1) * B + b1) * H + c] = (__bf16)acc1[v];
        }
    }
}

// ---------------------------------------------------------------------------
// Device-wide sense-reversing barrier (persistent kernel, all blocks resident)
// ---------------------------------------------------------------------------
DI void grid_barrier(int* bar, int& gen) {
    __syncthreads();
    if (threadIdx.x == 0) {
        __threadfence();
        int nb = (int)gridDim.x;
        int arrived = __hip_atomic_fetch_add(&bar[0], 1, __ATOMIC_ACQ_REL,
                                             __HIP_MEMORY_SCOPE_AGENT);
        if (arrived == nb - 1) {
            __hip_atomic_store(&bar[0], 0, __ATOMIC_RELAXED,
                               __HIP_MEMORY_SCOPE_AGENT);
            __hip_atomic_store(&bar[1], gen + 1, __ATOMIC_RELEASE,
                               __HIP_MEMORY_SCOPE_AGENT);
        } else {
            while (__hip_atomic_load(&bar[1], __ATOMIC_ACQUIRE,
                                     __HIP_MEMORY_SCOPE_AGENT) == gen) {
                __builtin_amdgcn_s_sleep(2);
            }
        }
    }
    gen += 1;
    __syncthreads();
}

// ---------------------------------------------------------------------------
// Phase 2: persistent recurrent scan. 64 waves; wave = one 16-column n-tile,
// all 3 gates, both M-tiles (6 accumulators, 192 WMMAs / layer-step).
// s kept as fp32 carry + bf16 WMMA shadow, both ping-pong. The bf16 state is
// staged into LDS per workgroup per layer-step via the async DMA path.
// ---------------------------------------------------------------------------
__global__ void __launch_bounds__(256, 1) k_rhn(const __bf16* __restrict__ U,
                                                const __bf16* __restrict__ Rb,
                                                const float* __restrict__ bh,
                                                const float* __restrict__ bt,
                                                const float* __restrict__ bc,
                                                float* __restrict__ s_a,
                                                float* __restrict__ s_b,
                                                __bf16* __restrict__ s_bf_a,
                                                __bf16* __restrict__ s_bf_b,
                                                int* __restrict__ bar,
                                                float* __restrict__ out) {
    __shared__ __bf16 s_lds[32 * PITCH];

    int lane = threadIdx.x & 31;
    int nt   = (int)blockIdx.x * 8 + (threadIdx.x >> 5);   // 0..63
    int c    = nt * 16 + (lane & 15);                      // output column
    int gen  = 0;

    const __bf16* a0r = s_lds + (size_t)(lane & 15) * PITCH;
    const __bf16* a1r = s_lds + (size_t)(16 + (lane & 15)) * PITCH;

    for (int t = 0; t < T; ++t) {
        for (int l = 0; l < L; ++l) {
            int step = t * L + l;
            const float*  s_cur    = (step & 1) ? s_b : s_a;
            float*        s_nxt    = (step & 1) ? s_a : s_b;
            const __bf16* s_cur_bf = (step & 1) ? s_bf_b : s_bf_a;
            __bf16*       s_nxt_bf = (step & 1) ? s_bf_a : s_bf_b;

            // Stage current bf16 state (32 x 1024, 64KB) into LDS.
            stage_rows_to_lds(s_lds, s_cur_bf, 32, H);

            const __bf16* rh = Rb + ((size_t)(l * 3 + 0) * H + c) * H;
            const __bf16* rt = Rb + ((size_t)(l * 3 + 1) * H + c) * H;
            const __bf16* rc = Rb + ((size_t)(l * 3 + 2) * H + c) * H;

            v8f ah0 = {}, ah1 = {}, at0 = {}, at1 = {}, ac0 = {}, ac1 = {};
#pragma unroll 2
            for (int k0 = 0; k0 < H; k0 += 32) {
                v16bf a0 = load_a_bf16(a0r, k0, lane);
                v16bf a1 = load_a_bf16(a1r, k0, lane);
                v16bf fh = load_b_bf16(rh, k0, lane);
                v16bf ft = load_b_bf16(rt, k0, lane);
                v16bf fc = load_b_bf16(rc, k0, lane);
                ah0 = wmma_bf16(a0, fh, ah0);
                ah1 = wmma_bf16(a1, fh, ah1);
                at0 = wmma_bf16(a0, ft, at0);
                at1 = wmma_bf16(a1, ft, at1);
                ac0 = wmma_bf16(a0, fc, ac0);
                ac1 = wmma_bf16(a1, fc, ac1);
            }

            float bhv = bh[l * H + c];
            float btv = bt[l * H + c];
            float bcv = bc[l * H + c];

#pragma unroll
            for (int mt = 0; mt < 2; ++mt) {
#pragma unroll
                for (int v = 0; v < 8; ++v) {
                    int b = mt * 16 + ((lane >> 4) & 1) * 8 + v;   // batch row
                    float hp = (mt ? ah1[v] : ah0[v]) + bhv;
                    float tp = (mt ? at1[v] : at0[v]) + btv;
                    float cp = (mt ? ac1[v] : ac0[v]) + bcv;
                    if (l == 0) {
                        hp += (float)U[(((size_t)0 * T + t) * B + b) * H + c];
                        tp += (float)U[(((size_t)1 * T + t) * B + b) * H + c];
                        cp += (float)U[(((size_t)2 * T + t) * B + b) * H + c];
                    }
                    float sold = s_cur[(size_t)b * H + c];
                    float hv = tanhf(hp);
                    float tv = sigmoidf_fast(tp);
                    float cv = sigmoidf_fast(cp);
                    float sn = hv * tv + sold * cv;
                    s_nxt[(size_t)b * H + c] = sn;
                    s_nxt_bf[(size_t)b * H + c] = (__bf16)sn;
                    if (l == L - 1) {
                        out[((size_t)b * T + t) * H + c] = sn;
                        if (t == T - 1)
                            out[(size_t)B * T * H + (size_t)b * H + c] = sn;
                    }
                }
            }
            grid_barrier(bar, gen);
        }
    }
}

// ---------------------------------------------------------------------------
// Host launcher
// ---------------------------------------------------------------------------
extern "C" void kernel_launch(void* const* d_in, const int* in_sizes, int n_in,
                              void* d_out, int out_size, void* d_ws, size_t ws_size,
                              hipStream_t stream) {
    (void)in_sizes; (void)n_in; (void)out_size; (void)ws_size;
    const float* x   = (const float*)d_in[0];
    const float* s0  = (const float*)d_in[1];
    const float* w_h = (const float*)d_in[2];
    const float* w_t = (const float*)d_in[3];
    const float* w_c = (const float*)d_in[4];
    const float* Rh  = (const float*)d_in[5];
    const float* Rt  = (const float*)d_in[6];
    const float* Rc  = (const float*)d_in[7];
    const float* bh  = (const float*)d_in[8];
    const float* bt  = (const float*)d_in[9];
    const float* bc  = (const float*)d_in[10];
    float* out = (float*)d_out;

    char* ws = (char*)d_ws;
    size_t off = 0;
    auto take = [&](size_t bytes) -> void* {
        void* p = ws + off;
        off = (off + bytes + 255) & ~(size_t)255;
        return p;
    };
    __bf16* U    = (__bf16*)take((size_t)3 * T * B * H * 2);   // 192 MiB
    __bf16* xb   = (__bf16*)take((size_t)B * T * I * 2);       //  64 MiB
    __bf16* wT   = (__bf16*)take((size_t)3 * H * I * 2);       //   6 MiB
    __bf16* Rb   = (__bf16*)take((size_t)L * 3 * H * H * 2);   //  18 MiB
    float*  s_a  = (float*)take((size_t)B * H * 4);
    float*  s_b  = (float*)take((size_t)B * H * 4);
    __bf16* sb_a = (__bf16*)take((size_t)B * H * 2);
    __bf16* sb_b = (__bf16*)take((size_t)B * H * 2);
    int*    bar  = (int*)take(256);

    k_prep_x<<<(B * T * I / 4 + 255) / 256, 256, 0, stream>>>(x, xb);
    k_prep_w<<<(3 * H * I + 255) / 256, 256, 0, stream>>>(w_h, w_t, w_c, wT);
    k_prep_R<<<(L * 3 * H * H + 255) / 256, 256, 0, stream>>>(Rh, Rt, Rc, Rb);
    k_prep_s<<<(B * H + 255) / 256, 256, 0, stream>>>(s0, s_a, sb_a, bar);

    k_proj<<<3 * ((B * T) / 32), 256, 0, stream>>>(xb, wT, U);

    k_rhn<<<RHN_BLOCKS, 256, 0, stream>>>(U, Rb, bh, bt, bc,
                                          s_a, s_b, sb_a, sb_b, bar, out);
}
